// EggRollLinear_65266323030652
// MI455X (gfx1250) — matrix operand order repaired and checked by reading, not compiled, not run on hardware
//
#include <hip/hip_runtime.h>

// ---------------------------------------------------------------------------
// EggRollLinear for MI455X (gfx1250, wave32, WMMA + Tensor Data Mover).
//
//   out = x @ W^T + bias + s * ((x @ (P @ b)) @ a^T)    s = 0.01/sqrt(16) = 0.0025
//
// Low-rank path re-associated (kills the 137-GFLOP P@(b@a^T) of the reference).
// Main 274-GFLOP GEMM: v_wmma_f32_16x16x32_bf16, 3-term bf16 hi/lo split
// (fp32-class accuracy at ~8/3 of the f32-WMMA instruction rate).
//
// Data movement per K-step (double-buffered, prefetch distance 1):
//   TDM tensor_load_to_lds (f32 tiles, TENSORcnt, pad -> 36-dword stride)
//   -> one-pass hi/lo split (packed dword VALU, ds_store_b64, done ONCE/elem)
//   -> bf16 tiles (80-byte stride) -> ds_load_b128 fragments -> 24 WMMAs.
// ---------------------------------------------------------------------------

typedef __attribute__((ext_vector_type(16))) __bf16        v16bf;
typedef __attribute__((ext_vector_type(8)))  __bf16        v8bf;
typedef __attribute__((ext_vector_type(8)))  float         v8f;
typedef __attribute__((ext_vector_type(4)))  float         f32x4;
typedef __attribute__((ext_vector_type(4)))  unsigned int  u32x4;
typedef __attribute__((ext_vector_type(2)))  unsigned int  u32x2;
typedef __attribute__((ext_vector_type(8)))  int           i32x8;
typedef __attribute__((ext_vector_type(4)))  int           i32x4;

#define MDIM 8192
#define NDIM 4096
#define KDIM 4096
#define RDIM 16

#define TM 128
#define TN 128
#define TK 32
#define NKSTEP (KDIM / TK)
#define STGW 36   // f32 staging row stride (dwords): 32 data + 4 TDM pad
#define LDT  40   // bf16 tile row stride (shorts): 80 B, 16B-aligned, conflict-free b128

// --- TDM D# builders (CDNA5 ISA) -------------------------
__device__ __forceinline__ u32x4 tdm_group0(unsigned lds_addr, unsigned long long gaddr) {
  u32x4 g;
  g[0] = 1u;                                                    // count=1 (valid), user mode
  g[1] = lds_addr;                                              // lds_addr[31:0] (bytes)
  g[2] = (unsigned)(gaddr & 0xffffffffu);                       // global_addr[31:0]
  g[3] = (unsigned)((gaddr >> 32) & 0x01ffffffu) | (2u << 30);  // global_addr[56:32] | type=2
  return g;
}

__device__ __forceinline__ i32x8 tdm_group1_f32tile(void) {
  // data_size=4B (2), pad_enable=1, pad_interval=32 dwords (code 4),
  // pad_amount=4 dwords (code 3). Tile 32 x 128 f32, tensor dims == tile dims,
  // row stride = KDIM elements.
  i32x8 g;
  g[0] = (2 << 16) | (1 << 20) | (4 << 22) | (3 << 25);
  g[1] = (int)((unsigned)TK << 16);          // tensor_dim0[15:0]=32
  g[2] = (int)((unsigned)TM << 16);          // tensor_dim0 hi | tensor_dim1 lo=128
  g[3] = (int)((unsigned)TK << 16);          // tensor_dim1 hi | tile_dim0=32
  g[4] = TM;                                 // tile_dim1=128 | tile_dim2=0
  g[5] = KDIM;                               // tensor_dim0_stride[31:0]
  g[6] = 0;
  g[7] = 0;
  return g;
}

__device__ __forceinline__ void tdm_issue(unsigned lds_addr, const float* gptr) {
  u32x4 g0 = tdm_group0(lds_addr, (unsigned long long)(size_t)gptr);
  i32x8 g1 = tdm_group1_f32tile();
  i32x4 z4 = {0, 0, 0, 0};
#if defined(__clang_major__) && (__clang_major__ >= 23)
  i32x8 z8 = {0, 0, 0, 0, 0, 0, 0, 0};
  __builtin_amdgcn_tensor_load_to_lds(g0, g1, z4, z4, z8, 0);
#else
  __builtin_amdgcn_tensor_load_to_lds(g0, g1, z4, z4, 0);
#endif
}

// --- one-pass hi/lo split of a float4, packed-dword stores (clean codegen) ---
// hi = truncate-to-bf16 (residual exact in f32); lo = round-to-nearest(residual)
__device__ __forceinline__ void split_store(f32x4 f, unsigned short* ph, unsigned short* pl) {
  unsigned int b0 = __float_as_uint(f.x) & 0xffff0000u;
  unsigned int b1 = __float_as_uint(f.y) & 0xffff0000u;
  unsigned int b2 = __float_as_uint(f.z) & 0xffff0000u;
  unsigned int b3 = __float_as_uint(f.w) & 0xffff0000u;
  float r0 = f.x - __uint_as_float(b0);
  float r1 = f.y - __uint_as_float(b1);
  float r2 = f.z - __uint_as_float(b2);
  float r3 = f.w - __uint_as_float(b3);
  u32x2 h, l;
  h.x = (b0 >> 16) | b1;
  h.y = (b2 >> 16) | b3;
  l.x = ((__float_as_uint(r0) + 0x8000u) >> 16) | ((__float_as_uint(r1) + 0x8000u) & 0xffff0000u);
  l.y = ((__float_as_uint(r2) + 0x8000u) >> 16) | ((__float_as_uint(r3) + 0x8000u) & 0xffff0000u);
  *(u32x2*)ph = h;
  *(u32x2*)pl = l;
}

__device__ __forceinline__ float dot4(f32x4 a, f32x4 b) {
  return a.x * b.x + a.y * b.y + a.z * b.z + a.w * b.w;
}

// ---------------------------------------------------------------------------
// t16[row,:] = A[row,:] @ B16   (B16 is K x 16 row-major); one WG per row.
// ---------------------------------------------------------------------------
__global__ __launch_bounds__(256) void rowdot16(const float* __restrict__ A,
                                                const float* __restrict__ B16,
                                                float* __restrict__ O16, int K) {
  __shared__ float red[16 * 256];
  const int row = blockIdx.x;
  const int t = threadIdx.x;
  const float* ar = A + (size_t)row * K;
  float acc[16];
#pragma unroll
  for (int r = 0; r < 16; ++r) acc[r] = 0.f;
  for (int k = t; k < K; k += 256) {
    const float av = ar[k];
    const f32x4* bp = (const f32x4*)(B16 + (size_t)k * 16);
    f32x4 b0 = bp[0], b1 = bp[1], b2 = bp[2], b3 = bp[3];
    acc[0]  = fmaf(av, b0.x, acc[0]);  acc[1]  = fmaf(av, b0.y, acc[1]);
    acc[2]  = fmaf(av, b0.z, acc[2]);  acc[3]  = fmaf(av, b0.w, acc[3]);
    acc[4]  = fmaf(av, b1.x, acc[4]);  acc[5]  = fmaf(av, b1.y, acc[5]);
    acc[6]  = fmaf(av, b1.z, acc[6]);  acc[7]  = fmaf(av, b1.w, acc[7]);
    acc[8]  = fmaf(av, b2.x, acc[8]);  acc[9]  = fmaf(av, b2.y, acc[9]);
    acc[10] = fmaf(av, b2.z, acc[10]); acc[11] = fmaf(av, b2.w, acc[11]);
    acc[12] = fmaf(av, b3.x, acc[12]); acc[13] = fmaf(av, b3.y, acc[13]);
    acc[14] = fmaf(av, b3.z, acc[14]); acc[15] = fmaf(av, b3.w, acc[15]);
  }
#pragma unroll
  for (int r = 0; r < 16; ++r) red[r * 256 + t] = acc[r];
  __syncthreads();
  for (int s = 128; s > 0; s >>= 1) {
    if (t < s) {
#pragma unroll
      for (int r = 0; r < 16; ++r) red[r * 256 + t] += red[r * 256 + t + s];
    }
    __syncthreads();
  }
  if (t < 16) O16[(size_t)row * 16 + t] = red[t * 256];
}

// ---------------------------------------------------------------------------
// Main GEMM: out[m,n] = sum_k x[m,k]*w[n,k] + bias[n] + 0.0025 * <t2[m,:], a[n,:]>
// 128x128x32 tiles; 8 waves (4 M x 2 N); TDM double-buffered f32 staging.
// ---------------------------------------------------------------------------
__global__ __launch_bounds__(256) void egg_gemm(const float* __restrict__ x,
                                                const float* __restrict__ w,
                                                const float* __restrict__ bias,
                                                const float* __restrict__ noise,
                                                const float* __restrict__ t2,
                                                float* __restrict__ out) {
  __shared__ __align__(16) float smStgA[2][TM * STGW];        // 2 x 18 KB (TDM dest)
  __shared__ __align__(16) float smStgB[2][TN * STGW];        // 2 x 18 KB (TDM dest)
  __shared__ __align__(16) unsigned short smAh[TM * LDT];     // 10 KB each
  __shared__ __align__(16) unsigned short smAl[TM * LDT];
  __shared__ __align__(16) unsigned short smBh[TN * LDT];
  __shared__ __align__(16) unsigned short smBl[TN * LDT];

  const int tid   = threadIdx.x;
  const int m0    = blockIdx.y * TM;
  const int n0    = blockIdx.x * TN;
  const int lane  = tid & 31;
  const int wv    = tid >> 5;
  const int wM    = wv & 3;     // 4 waves along M (32 rows each)
  const int wN    = wv >> 2;    // 2 waves along N (64 cols each)
  const int laneM = lane & 15;
  const int laneH = lane >> 4;

  const int r0 = tid >> 3;        // conversion row base (0..31), +i*32
  const int c4 = (tid & 7) * 4;   // conversion k offset (elements)

  const v8f vzero = {0.f, 0.f, 0.f, 0.f, 0.f, 0.f, 0.f, 0.f};
  v8f acc[2][4];
#pragma unroll
  for (int mi = 0; mi < 2; ++mi)
#pragma unroll
    for (int ni = 0; ni < 4; ++ni) acc[mi][ni] = vzero;

  const unsigned ldsA[2] = {(unsigned)(size_t)&smStgA[0][0], (unsigned)(size_t)&smStgA[1][0]};
  const unsigned ldsB[2] = {(unsigned)(size_t)&smStgB[0][0], (unsigned)(size_t)&smStgB[1][0]};
  const float* gA = x + (size_t)m0 * KDIM;
  const float* gB = w + (size_t)n0 * KDIM;

  // Prologue: wave 0 issues the first K-slab (TDM ignores EXEC; one issue/wave).
  if (tid < 32) {
    tdm_issue(ldsA[0], gA);
    tdm_issue(ldsB[0], gB);
  }

  for (int kt = 0; kt < NKSTEP; ++kt) {
    const int cur = kt & 1;
    if (tid < 32) {
      if (kt + 1 < NKSTEP) {
        const size_t knext = (size_t)(kt + 1) * TK;
        tdm_issue(ldsA[cur ^ 1], gA + knext);   // buffer free since kt-1's 2nd barrier
        tdm_issue(ldsB[cur ^ 1], gB + knext);
        __builtin_amdgcn_s_wait_tensorcnt(2);   // in-order: pair(kt) complete
      } else {
        __builtin_amdgcn_s_wait_tensorcnt(0);
      }
    }
    __syncthreads();  // staging[cur] ready; bf16 tiles free (WMMA kt-1 ds-reads fenced)

    // --- one-pass conversion: staging f32 -> packed bf16 hi/lo tiles ---------
    const float* sa = smStgA[cur];
    const float* sb = smStgB[cur];
#pragma unroll
    for (int i = 0; i < 4; ++i) {
      const int row = r0 + i * 32;
      f32x4 fa = *(const f32x4*)(sa + row * STGW + c4);
      split_store(fa, &smAh[row * LDT + c4], &smAl[row * LDT + c4]);
      f32x4 fb = *(const f32x4*)(sb + row * STGW + c4);
      split_store(fb, &smBh[row * LDT + c4], &smBl[row * LDT + c4]);
    }
    __syncthreads();  // bf16 tiles ready

    // --- A fragments (ISA 16-bit A 16x32 layout: lane=M; two 16B K-chunks) ---
    v16bf ah[2], al[2];
#pragma unroll
    for (int mi = 0; mi < 2; ++mi) {
      const unsigned short* pa = &smAh[(wM * 32 + mi * 16 + laneM) * LDT + laneH * 8];
      const unsigned short* qa = &smAl[(wM * 32 + mi * 16 + laneM) * LDT + laneH * 8];
      v8bf a0 = *(const v8bf*)pa;
      v8bf a1 = *(const v8bf*)(pa + 16);
      ah[mi] = __builtin_shufflevector(a0, a1, 0,1,2,3,4,5,6,7,8,9,10,11,12,13,14,15);
      v8bf c0 = *(const v8bf*)qa;
      v8bf c1 = *(const v8bf*)(qa + 16);
      al[mi] = __builtin_shufflevector(c0, c1, 0,1,2,3,4,5,6,7,8,9,10,11,12,13,14,15);
    }

    // --- B fragments (lane=N column; two 8-K chunks, same K order as A) + WMMA --------
#pragma unroll
    for (int ni = 0; ni < 4; ++ni) {
      const unsigned short* pb = &smBh[(wN * 64 + ni * 16 + laneM) * LDT + laneH * 8];
      const unsigned short* qb = &smBl[(wN * 64 + ni * 16 + laneM) * LDT + laneH * 8];
      v8bf b0 = *(const v8bf*)pb;
      v8bf b1 = *(const v8bf*)(pb + 16);
      v16bf bh = __builtin_shufflevector(b0, b1, 0,1,2,3,4,5,6,7,8,9,10,11,12,13,14,15);
      v8bf d0 = *(const v8bf*)qb;
      v8bf d1 = *(const v8bf*)(qb + 16);
      v16bf bl = __builtin_shufflevector(d0, d1, 0,1,2,3,4,5,6,7,8,9,10,11,12,13,14,15);
#pragma unroll
      for (int mi = 0; mi < 2; ++mi) {
        acc[mi][ni] = __builtin_amdgcn_wmma_f32_16x16x32_bf16(
            false, ah[mi], false, bh, (short)0, acc[mi][ni], false, false);
        acc[mi][ni] = __builtin_amdgcn_wmma_f32_16x16x32_bf16(
            false, ah[mi], false, bl, (short)0, acc[mi][ni], false, false);
        acc[mi][ni] = __builtin_amdgcn_wmma_f32_16x16x32_bf16(
            false, al[mi], false, bh, (short)0, acc[mi][ni], false, false);
      }
    }
  }

  // --- epilogue: bias + rank-16 delta; C layout: lane = N, VGPR v -> M row ---
  const float scale = 0.0025f;                       // SIGMA/sqrt(16)*SIGN
  const float* amat = noise + (size_t)KDIM * RDIM;   // a_mat = noise[in_f:,:]
  f32x4 AR[4][4]; float BN[4]; int NN[4];
#pragma unroll
  for (int ni = 0; ni < 4; ++ni) {
    const int n = n0 + wN * 64 + ni * 16 + laneM;
    NN[ni] = n;
    BN[ni] = bias[n];
    const f32x4* ap = (const f32x4*)(amat + (size_t)n * RDIM);
    AR[ni][0] = ap[0]; AR[ni][1] = ap[1]; AR[ni][2] = ap[2]; AR[ni][3] = ap[3];
  }
#pragma unroll
  for (int mi = 0; mi < 2; ++mi) {
#pragma unroll
    for (int v = 0; v < 8; ++v) {
      const int m = m0 + wM * 32 + mi * 16 + laneH * 8 + v;
      const f32x4* tp = (const f32x4*)(t2 + (size_t)m * RDIM);
      f32x4 t0v = tp[0], t1v = tp[1], t2v = tp[2], t3v = tp[3];
#pragma unroll
      for (int ni = 0; ni < 4; ++ni) {
        const float d = dot4(t0v, AR[ni][0]) + dot4(t1v, AR[ni][1]) +
                        dot4(t2v, AR[ni][2]) + dot4(t3v, AR[ni][3]);
        out[(size_t)m * NDIM + NN[ni]] = acc[mi][ni][v] + BN[ni] + scale * d;
      }
    }
  }
}

extern "C" void kernel_launch(void* const* d_in, const int* in_sizes, int n_in,
                              void* d_out, int out_size, void* d_ws, size_t ws_size,
                              hipStream_t stream) {
  const float* x      = (const float*)d_in[0];  // (8192, 4096)
  const float* weight = (const float*)d_in[1];  // (4096, 4096)
  const float* bias   = (const float*)d_in[2];  // (4096,)
  const float* noise  = (const float*)d_in[3];  // (8192, 16)
  const float* P      = (const float*)d_in[4];  // (4096, 4096)
  float* out = (float*)d_out;

  float* t1 = (float*)d_ws;            // (4096, 16) = P @ b_mat
  float* t2 = t1 + (size_t)KDIM * 16;  // (8192, 16) = x @ t1   (768 KB ws total)

  rowdot16<<<KDIM, 256, 0, stream>>>(P, noise, t1, KDIM);
  rowdot16<<<MDIM, 256, 0, stream>>>(x, t1, t2, KDIM);

  dim3 grid(NDIM / TN, MDIM / TM);  // 32 x 64
  egg_gemm<<<grid, 256, 0, stream>>>(x, weight, bias, noise, t2, out);
}
